// Model_58360015618168
// MI455X (gfx1250) — matrix operand order repaired and checked by reading, not compiled, not run on hardware
//
#include <hip/hip_runtime.h>
#include <hip/hip_bf16.h>

typedef __bf16 bf16_t;
typedef float  f32x8  __attribute__((ext_vector_type(8)));
typedef bf16_t bf16x8 __attribute__((ext_vector_type(8)));
typedef bf16_t bf16x16 __attribute__((ext_vector_type(16)));

#define BB   8
#define SS   1024
#define DD   512
#define LLAY 4
#define NH   8
#define DH   64
#define FF   2048
#define MTOT (BB * SS)

// ---------------- WMMA helpers ----------------
static __device__ __forceinline__ f32x8 wmma_bf16(bf16x16 a, bf16x16 b, f32x8 c) {
  return __builtin_amdgcn_wmma_f32_16x16x32_bf16(false, a, false, b, (short)0, c, false, false);
}
static __device__ __forceinline__ bf16x16 cat8(bf16x8 lo, bf16x8 hi) {
  return __builtin_shufflevector(lo, hi, 0,1,2,3,4,5,6,7,8,9,10,11,12,13,14,15);
}
static __device__ __forceinline__ float wave_sum(float v) {
#pragma unroll
  for (int o = 16; o > 0; o >>= 1) v += __shfl_xor(v, o, 32);
  return v;
}

// ---------------- positional embedding: h = x + pe ----------------
__global__ __launch_bounds__(256) void k_posemb(const float* __restrict__ x,
                                                float* __restrict__ h) {
  int idx = blockIdx.x * 256 + threadIdx.x;
  if (idx >= BB * SS * DD) return;
  int d = idx & (DD - 1);
  int s = (idx >> 9) & (SS - 1);
  float expo = (2.0f * (float)d) / (float)DD;
  float ang  = (float)s * __expf(-expo * 9.2103403719761836f); // ln(10000)
  float pe   = (d < DD / 2) ? __sinf(ang) : __cosf(ang);
  h[idx] = x[idx] + pe;
}

// ---------------- convert+transpose weight: Wt[n,k] = bf16(W[k,n]) ----------------
__global__ __launch_bounds__(256) void k_cvtT(const float* __restrict__ W,
                                              bf16_t* __restrict__ Wt, int K, int N) {
  int idx = blockIdx.x * 256 + threadIdx.x;
  if (idx >= K * N) return;
  int n = idx / K;
  int k = idx - n * K;
  Wt[idx] = (bf16_t)W[(size_t)k * N + n];
}

// ---------------- layernorm (D=512) -> bf16, one wave per row ----------------
__global__ __launch_bounds__(256) void k_ln(const float* __restrict__ h,
                                            const float* __restrict__ g,
                                            const float* __restrict__ b,
                                            bf16_t* __restrict__ out) {
  int wv = threadIdx.x >> 5, lane = threadIdx.x & 31;
  int row = blockIdx.x * 8 + wv;
  const float* hr = h + (size_t)row * DD;
  float xr[16], s = 0.f;
#pragma unroll
  for (int t = 0; t < 16; ++t) { xr[t] = hr[lane + 32 * t]; s += xr[t]; }
  float mu = wave_sum(s) * (1.0f / DD);
  float v = 0.f;
#pragma unroll
  for (int t = 0; t < 16; ++t) { float d = xr[t] - mu; v += d * d; }
  float inv = rsqrtf(wave_sum(v) * (1.0f / DD) + 1e-3f);
  bf16_t* orow = out + (size_t)row * DD;
#pragma unroll
  for (int t = 0; t < 16; ++t) {
    int c = lane + 32 * t;
    orow[c] = (bf16_t)((xr[t] - mu) * inv * g[c] + b[c]);
  }
}

// ---------------- generic bf16 WMMA GEMM: C[M,N] = A[M,K] @ Bt[N,K]^T ----------------
// wave computes 32x64 (2 M-subtiles, B fragments reused); block = 8 waves -> 256x64.
// epilogue: +bias, relu, +resid(f32), write fp32 and/or bf16.
__global__ __launch_bounds__(256)
void k_gemm(const bf16_t* __restrict__ A, const bf16_t* __restrict__ Bt,
            const float* __restrict__ bias, const float* __restrict__ resid,
            float* __restrict__ outF, bf16_t* __restrict__ outB,
            int M, int N, int K, int relu) {
  int lane = threadIdx.x & 31;
  int wv   = threadIdx.x >> 5;
  int hf   = lane >> 4, l15 = lane & 15;
  int n0   = blockIdx.x * 64;
  int m0   = blockIdx.y * 256 + wv * 32;

  const bf16_t* ap0 = A  + (size_t)(m0 + l15) * K;
  const bf16_t* ap1 = A  + (size_t)(m0 + 16 + l15) * K;
  const bf16_t* bp0 = Bt + (size_t)(n0 + l15) * K + hf * 16;

  f32x8 acc[2][4] = {};
  for (int k = 0; k < K; k += 32) {
    bf16x16 a0 = cat8(*(const bf16x8*)(ap0 + k + hf * 16),
                      *(const bf16x8*)(ap0 + k + hf * 16 + 8));
    bf16x16 a1 = cat8(*(const bf16x8*)(ap1 + k + hf * 16),
                      *(const bf16x8*)(ap1 + k + hf * 16 + 8));
#pragma unroll
    for (int j = 0; j < 4; ++j) {
      bf16x16 bfrag = *(const bf16x16*)(bp0 + (size_t)j * 16 * K + k);
      acc[0][j] = wmma_bf16(a0, bfrag, acc[0][j]);
      acc[1][j] = wmma_bf16(a1, bfrag, acc[1][j]);
    }
  }
#pragma unroll
  for (int t = 0; t < 2; ++t) {
#pragma unroll
    for (int j = 0; j < 4; ++j) {
      int col = n0 + j * 16 + l15;
      float bv = bias ? bias[col] : 0.f;
#pragma unroll
      for (int r = 0; r < 8; ++r) {
        int row = m0 + t * 16 + hf * 8 + r;
        float v = acc[t][j][r] + bv;
        if (relu) v = fmaxf(v, 0.f);
        size_t o = (size_t)row * N + col;
        if (resid) v += resid[o];
        if (outF) outF[o] = v;
        if (outB) outB[o] = (bf16_t)v;
      }
    }
  }
}

// ---------------- flash attention: one wave per (b, h, 32-query tile) ----------------
// 2 query subtiles share K and V fragments; softmax is fully wave32-parallel (row = lane).
__global__ __launch_bounds__(32)
void k_attn(const bf16_t* __restrict__ Q, const bf16_t* __restrict__ Km,
            const bf16_t* __restrict__ V, const unsigned char* __restrict__ mask,
            bf16_t* __restrict__ O) {
  int lane = threadIdx.x;
  int qt = blockIdx.x, hh = blockIdx.y, bb = blockIdx.z;
  int hoff = hh * DH;
  int qrow_g = bb * SS + qt * 32;
  int hf = lane >> 4, l15 = lane & 15;

  __shared__ __align__(32) float  sbuf[32][33];
  __shared__ __align__(32) bf16_t pbuf[32][32];
  __shared__ float afac[32];

  // Q fragments: 2 query subtiles x 2 dh-chunks (dh=64 -> K split 32+32)
  bf16x16 qa[2][2];
#pragma unroll
  for (int t = 0; t < 2; ++t) {
    const bf16_t* qp = Q + (size_t)(qrow_g + t * 16 + l15) * DD + hoff;
    qa[t][0] = cat8(*(const bf16x8*)(qp + hf * 16),      *(const bf16x8*)(qp + hf * 16 + 8));
    qa[t][1] = cat8(*(const bf16x8*)(qp + 32 + hf * 16), *(const bf16x8*)(qp + 40 + hf * 16));
  }

  f32x8 o[2][4] = {};
  float mrun = -1e30f, lrun = 0.f;  // per-lane: row = lane
  const unsigned char* mrow = mask + (size_t)bb * SS;

  for (int kb = 0; kb < SS; kb += 32) {
#pragma unroll
    for (int kt = 0; kt < 2; ++kt) {
      int key = kb + kt * 16 + l15;
      const bf16_t* kp = Km + (size_t)(bb * SS + key) * DD + hoff + hf * 16;
      bf16x16 kf0 = *(const bf16x16*)(kp);
      bf16x16 kf1 = *(const bf16x16*)(kp + 32);
      bool valid = mrow[key] != 0;
#pragma unroll
      for (int t = 0; t < 2; ++t) {
        f32x8 c = {};
        c = wmma_bf16(qa[t][0], kf0, c);
        c = wmma_bf16(qa[t][1], kf1, c);
#pragma unroll
        for (int r = 0; r < 8; ++r)
          sbuf[t * 16 + hf * 8 + r][kt * 16 + l15] = valid ? c[r] * 0.125f : -1e9f;
      }
    }
    __syncthreads();
    {
      // softmax bookkeeping: lane owns row = lane (all 32 lanes active)
      float sv[32], tm = -1e30f;
#pragma unroll
      for (int j = 0; j < 32; ++j) { sv[j] = sbuf[lane][j]; tm = fmaxf(tm, sv[j]); }
      float nm = fmaxf(mrun, tm);
      float al = __expf(mrun - nm);
      float ps = 0.f;
#pragma unroll
      for (int j = 0; j < 32; ++j) {
        float pp = __expf(sv[j] - nm);
        ps += pp;
        pbuf[lane][j] = (bf16_t)pp;
      }
      lrun = lrun * al + ps;
      mrun = nm;
      afac[lane] = al;
    }
    __syncthreads();
#pragma unroll
    for (int t = 0; t < 2; ++t)
#pragma unroll
      for (int r = 0; r < 8; ++r) {
        float al = afac[t * 16 + hf * 8 + r];
#pragma unroll
        for (int j = 0; j < 4; ++j) o[t][j][r] *= al;
      }
    // P as A fragments (from LDS), one per query subtile
    bf16x16 pa0 = cat8(*(const bf16x8*)&pbuf[l15][hf * 16],
                       *(const bf16x8*)&pbuf[l15][hf * 16 + 8]);
    bf16x16 pa1 = cat8(*(const bf16x8*)&pbuf[16 + l15][hf * 16],
                       *(const bf16x8*)&pbuf[16 + l15][hf * 16 + 8]);
    // V fragments gathered once, reused for both query subtiles
#pragma unroll
    for (int j = 0; j < 4; ++j) {
      bf16x16 vb;
#pragma unroll
      for (int t = 0; t < 16; ++t)
        vb[t] = V[(size_t)(bb * SS + kb + hf * 16 + t) * DD + hoff + j * 16 + l15];
      o[0][j] = wmma_bf16(pa0, vb, o[0][j]);
      o[1][j] = wmma_bf16(pa1, vb, o[1][j]);
    }
    __syncthreads();
  }
  afac[lane] = 1.0f / lrun;
  __syncthreads();
#pragma unroll
  for (int t = 0; t < 2; ++t) {
    float li[8];
#pragma unroll
    for (int r = 0; r < 8; ++r) li[r] = afac[t * 16 + hf * 8 + r];
#pragma unroll
    for (int j = 0; j < 4; ++j)
#pragma unroll
      for (int r = 0; r < 8; ++r)
        O[(size_t)(qrow_g + t * 16 + hf * 8 + r) * DD + hoff + j * 16 + l15] =
            (bf16_t)(o[t][j][r] * li[r]);
  }
}

// ---------------- host orchestration ----------------
extern "C" void kernel_launch(void* const* d_in, const int* in_sizes, int n_in,
                              void* d_out, int out_size, void* d_ws, size_t ws_size,
                              hipStream_t stream) {
  (void)in_sizes; (void)n_in; (void)out_size; (void)ws_size;
  const float* x   = (const float*)d_in[0];
  const unsigned char* mask = (const unsigned char*)d_in[1];
  const float* Wq  = (const float*)d_in[2];
  const float* Wk  = (const float*)d_in[3];
  const float* Wv  = (const float*)d_in[4];
  const float* Wo  = (const float*)d_in[5];
  const float* W1  = (const float*)d_in[6];
  const float* b1  = (const float*)d_in[7];
  const float* W2  = (const float*)d_in[8];
  const float* b2  = (const float*)d_in[9];
  const float* g1v = (const float*)d_in[10];
  const float* b1v = (const float*)d_in[11];
  const float* g2v = (const float*)d_in[12];
  const float* b2v = (const float*)d_in[13];

  float* h = (float*)d_out; // residual stream lives in d_out

  char* p = (char*)d_ws;
  auto carve = [&](size_t bytes) -> void* {
    void* r = (void*)p;
    p += (bytes + 255) & ~(size_t)255;
    return r;
  };
  bf16_t* hn   = (bf16_t*)carve((size_t)MTOT * DD * 2);
  bf16_t* qb   = (bf16_t*)carve((size_t)MTOT * DD * 2);
  bf16_t* kbuf = (bf16_t*)carve((size_t)MTOT * DD * 2);
  bf16_t* vbuf = (bf16_t*)carve((size_t)MTOT * DD * 2);
  bf16_t* ab   = (bf16_t*)carve((size_t)MTOT * DD * 2);
  bf16_t* f1   = (bf16_t*)carve((size_t)MTOT * FF * 2);
  bf16_t* Wqt  = (bf16_t*)carve((size_t)DD * DD * 2);
  bf16_t* Wkt  = (bf16_t*)carve((size_t)DD * DD * 2);
  bf16_t* Wvt  = (bf16_t*)carve((size_t)DD * DD * 2);
  bf16_t* Wot  = (bf16_t*)carve((size_t)DD * DD * 2);
  bf16_t* W1t  = (bf16_t*)carve((size_t)DD * FF * 2);
  bf16_t* W2t  = (bf16_t*)carve((size_t)FF * DD * 2);

  k_posemb<<<(MTOT * DD) / 256, 256, 0, stream>>>(x, h);

  dim3 gD(DD / 64, MTOT / 256);   // N=512 GEMMs
  dim3 gF(FF / 64, MTOT / 256);   // N=2048 GEMM
  dim3 gA(SS / 32, NH, BB);       // attention

  for (int i = 0; i < LLAY; ++i) {
    k_cvtT<<<(DD * DD + 255) / 256, 256, 0, stream>>>(Wq + (size_t)i * DD * DD, Wqt, DD, DD);
    k_cvtT<<<(DD * DD + 255) / 256, 256, 0, stream>>>(Wk + (size_t)i * DD * DD, Wkt, DD, DD);
    k_cvtT<<<(DD * DD + 255) / 256, 256, 0, stream>>>(Wv + (size_t)i * DD * DD, Wvt, DD, DD);
    k_cvtT<<<(DD * DD + 255) / 256, 256, 0, stream>>>(Wo + (size_t)i * DD * DD, Wot, DD, DD);
    k_cvtT<<<(DD * FF + 255) / 256, 256, 0, stream>>>(W1 + (size_t)i * DD * FF, W1t, DD, FF);
    k_cvtT<<<(FF * DD + 255) / 256, 256, 0, stream>>>(W2 + (size_t)i * FF * DD, W2t, FF, DD);

    k_ln<<<MTOT / 8, 256, 0, stream>>>(h, g1v + i * DD, b1v + i * DD, hn);
    k_gemm<<<gD, 256, 0, stream>>>(hn, Wqt, nullptr, nullptr, nullptr, qb,   MTOT, DD, DD, 0);
    k_gemm<<<gD, 256, 0, stream>>>(hn, Wkt, nullptr, nullptr, nullptr, kbuf, MTOT, DD, DD, 0);
    k_gemm<<<gD, 256, 0, stream>>>(hn, Wvt, nullptr, nullptr, nullptr, vbuf, MTOT, DD, DD, 0);
    k_attn<<<gA, 32, 0, stream>>>(qb, kbuf, vbuf, mask, ab);
    k_gemm<<<gD, 256, 0, stream>>>(ab, Wot, nullptr, h, h, nullptr, MTOT, DD, DD, 0);

    k_ln<<<MTOT / 8, 256, 0, stream>>>(h, g2v + i * DD, b2v + i * DD, hn);
    k_gemm<<<gF, 256, 0, stream>>>(hn, W1t, b1 + (size_t)i * FF, nullptr, nullptr, f1, MTOT, FF, DD, 1);
    k_gemm<<<gD, 256, 0, stream>>>(f1, W2t, b2 + (size_t)i * DD, h, h, nullptr, MTOT, DD, FF, 0);
  }
}
